// GraphCovLayer_69483980914743
// MI455X (gfx1250) — matrix operand order repaired and checked by reading, not compiled, not run on hardware
//
#include <hip/hip_runtime.h>
#include <stdint.h>

typedef __attribute__((ext_vector_type(16))) _Float16 v16h;
typedef __attribute__((ext_vector_type(8)))  _Float16 v8h;
typedef __attribute__((ext_vector_type(2)))  _Float16 v2h;
typedef __attribute__((ext_vector_type(8)))  float    v8f;

#define R_CAT 5
#define FEAT  64

// ---------------- utility: zero buffers ----------------
__global__ void zero_f32_kernel(float* __restrict__ p, size_t n) {
    size_t i = (size_t)blockIdx.x * blockDim.x + threadIdx.x;
    size_t stride = (size_t)gridDim.x * blockDim.x;
    for (; i < n; i += stride) p[i] = 0.0f;
}

__global__ void zero_u32_kernel(unsigned int* __restrict__ p, size_t n) {
    size_t i = (size_t)blockIdx.x * blockDim.x + threadIdx.x;
    size_t stride = (size_t)gridDim.x * blockDim.x;
    for (; i < n; i += stride) p[i] = 0u;
}

// ---------------- weight prep: f32 [r][i][o] -> f16 [r][o][i] ----------------
// Row o of wT[r] is the WMMA A-operand M-row (K = i contiguous).
__global__ void prep_weight_kernel(const float* __restrict__ w, _Float16* __restrict__ wT) {
    int idx = blockIdx.x * blockDim.x + threadIdx.x;     // r*4096 + i*64 + o
    if (idx >= R_CAT * FEAT * FEAT) return;
    int r   = idx / (FEAT * FEAT);
    int rem = idx % (FEAT * FEAT);
    int i   = rem / FEAT;
    int o   = rem % FEAT;
    wT[((size_t)r * FEAT + o) * FEAT + i] = (_Float16)w[idx];
}

// ---------------- per-(dst,rate) edge counts ----------------
__global__ void count_edges_kernel(const int* __restrict__ dst, const int* __restrict__ rate,
                                   unsigned int* __restrict__ cnt, int nE) {
    int e = blockIdx.x * blockDim.x + threadIdx.x;
    if (e < nE) atomicAdd(&cnt[(size_t)dst[e] * R_CAT + rate[e]], 1u);
}

// ---------------- node transform (transposed WMMA): D[o][node] = W^T[r] x^T ----------------
// One block = 16 node columns (N). 4 wave32's, each owns a 16-row o-tile (M).
//   A = wT[r] tile (M=o, K=i)  : direct contiguous f16 loads, no conversion
//   B = x^T      (K=i, N=node) : 2x8 consecutive floats/lane per step -> cvt_pk f16, reused for all r
//   D : lane = node, VGPR j = consecutive o  -> one packed b128 store per lane per r
__global__ __launch_bounds__(128)
void node_transform_kernel(const float* __restrict__ x, const _Float16* __restrict__ wT,
                           _Float16* __restrict__ xw, int n_nodes) {
    const int lane  = threadIdx.x & 31;
    const int wave  = threadIdx.x >> 5;         // 0..3 -> 16-row o tile
    const int node16 = blockIdx.x * 16;
    const int lo    = lane & 15;
    const int hi    = lane >> 4;                // 0 or 1

    // ----- B operand (64x16): lane holds K = hi*8 + {0..7, 16..23} per 32-K step
    int node = node16 + lo;
    const bool nodeValid = node < n_nodes;
    if (!nodeValid) node = n_nodes - 1;         // clamp loads; WMMA needs full EXEC
    const float* bp = x + (size_t)node * FEAT + hi * 8;
    v8f g0 = *(const v8f*)(bp +  0);            // K = kb     .. kb+7   (kb = hi*8)
    v8f g1 = *(const v8f*)(bp + 16);            // K = 16+kb  .. +7
    v8f g2 = *(const v8f*)(bp + 32);            // K = 32+kb  .. +7
    v8f g3 = *(const v8f*)(bp + 48);            // K = 48+kb  .. +7
    v16h b0, b1;
#pragma unroll
    for (int i = 0; i < 8; ++i) {
        b0[i]     = (_Float16)g0[i];
        b0[8 + i] = (_Float16)g1[i];
        b1[i]     = (_Float16)g2[i];
        b1[8 + i] = (_Float16)g3[i];
    }

    const int orow = wave * 16 + lo;            // A-operand M row this lane feeds

#pragma unroll
    for (int r = 0; r < R_CAT; ++r) {
        // ----- A operand (16x32 f16 x2): lane holds K = hi*8 + {0..7, 16..23} per step
        const _Float16* ap = wT + ((size_t)(r * FEAT + orow)) * FEAT + hi * 8;
        v8h t0 = *(const v8h*)(ap +  0);
        v8h t1 = *(const v8h*)(ap + 16);
        v8h t2 = *(const v8h*)(ap + 32);
        v8h t3 = *(const v8h*)(ap + 48);
        v16h a0, a1;
#pragma unroll
        for (int i = 0; i < 8; ++i) {
            a0[i] = t0[i]; a0[8 + i] = t1[i];
            a1[i] = t2[i]; a1[8 + i] = t3[i];
        }

        v8f c = {};
        c = __builtin_amdgcn_wmma_f32_16x16x32_f16(false, a0, false, b0,
                                                   (short)0, c, false, false);
        c = __builtin_amdgcn_wmma_f32_16x16x32_f16(false, a1, false, b1,
                                                   (short)0, c, false, false);

        // ----- D: lane(lo) = node col; VGPR j -> o = wave*16 + hi*8 + j (8 consecutive)
        v8h d;
#pragma unroll
        for (int j = 0; j < 8; ++j) d[j] = (_Float16)c[j];
        if (nodeValid) {
            _Float16* op = xw + ((size_t)node * R_CAT + r) * FEAT + wave * 16 + hi * 8;
            *(v8h*)op = d;                       // one global_store_b128
        }
    }
}

// ---------------- edge scatter: out[dst] += xw[src][rate] / cnt[dst*R+rate] ----------------
// One wave per edge; each lane owns 2 of the 64 features (coalesced 128B gather).
__global__ __launch_bounds__(256)
void scatter_edges_kernel(const _Float16* __restrict__ xw_src,
                          const int* __restrict__ src_idx, const int* __restrict__ dst_idx,
                          const int* __restrict__ rate, const unsigned int* __restrict__ cnt_dst,
                          float* __restrict__ out, int nE) {
    const int lane = threadIdx.x & 31;
    const int wavesPerBlock = blockDim.x >> 5;
    int e = blockIdx.x * wavesPerBlock + (threadIdx.x >> 5);
    if (e >= nE) return;

    const int s = src_idx[e];
    const int d = dst_idx[e];
    const int r = rate[e];
    const float inv = 1.0f / (float)cnt_dst[(size_t)d * R_CAT + r];

    const v2h mv = *(const v2h*)(xw_src + (((size_t)s * R_CAT + r) * FEAT) + lane * 2);
    float m0 = (float)mv[0] * inv;
    float m1 = (float)mv[1] * inv;

    float* op = out + (size_t)d * FEAT + lane * 2;
    atomicAdd(op,     m0);
    atomicAdd(op + 1, m1);
}

// ---------------- host side ----------------
extern "C" void kernel_launch(void* const* d_in, const int* in_sizes, int n_in,
                              void* d_out, int out_size, void* d_ws, size_t ws_size,
                              hipStream_t stream) {
    const float* user_x = (const float*)d_in[0];
    const float* item_x = (const float*)d_in[1];
    const float* weight = (const float*)d_in[2];
    const int*   u_s    = (const int*)d_in[3];
    const int*   v_s    = (const int*)d_in[4];
    const int*   rate   = (const int*)d_in[5];

    const int nU = in_sizes[0] / FEAT;
    const int nI = in_sizes[1] / FEAT;
    const int nE = in_sizes[3];

    float* h_u = (float*)d_out;                       // [nU, 64]
    float* h_v = (float*)d_out + (size_t)nU * FEAT;   // [nI, 64]

    // workspace carve-out (256B aligned slots):
    uint8_t* ws = (uint8_t*)d_ws;
    size_t off = 0;
    auto carve = [&](size_t bytes) -> uint8_t* {
        uint8_t* p = ws + off;
        off = (off + bytes + 255) & ~(size_t)255;
        return p;
    };
    _Float16*     xw_u  = (_Float16*)carve((size_t)nU * R_CAT * FEAT * sizeof(_Float16)); // 64 MB
    _Float16*     xw_i  = (_Float16*)carve((size_t)nI * R_CAT * FEAT * sizeof(_Float16)); // 32 MB
    unsigned int* cnt_u = (unsigned int*)carve((size_t)nU * R_CAT * sizeof(unsigned int)); // 2 MB
    unsigned int* cnt_i = (unsigned int*)carve((size_t)nI * R_CAT * sizeof(unsigned int)); // 1 MB
    _Float16*     wT    = (_Float16*)carve((size_t)R_CAT * FEAT * FEAT * sizeof(_Float16));

    // 1) zero accumulators (d_out / counts are poisoned or stale)
    zero_f32_kernel<<<2048, 256, 0, stream>>>((float*)d_out, (size_t)(nU + nI) * FEAT);
    zero_u32_kernel<<<1024, 256, 0, stream>>>(cnt_u, (size_t)nU * R_CAT);
    zero_u32_kernel<<<512,  256, 0, stream>>>(cnt_i, (size_t)nI * R_CAT);

    // 2) weight prep (transpose + f16 convert, reused by ~9k GEMM blocks)
    prep_weight_kernel<<<(R_CAT * FEAT * FEAT + 255) / 256, 256, 0, stream>>>(weight, wT);

    // 3) per-(dst,rate) edge counts
    count_edges_kernel<<<(nE + 255) / 256, 256, 0, stream>>>(u_s, rate, cnt_u, nE);
    count_edges_kernel<<<(nE + 255) / 256, 256, 0, stream>>>(v_s, rate, cnt_i, nE);

    // 4) WMMA node transforms
    node_transform_kernel<<<(nU + 15) / 16, 128, 0, stream>>>(user_x, wT, xw_u, nU);
    node_transform_kernel<<<(nI + 15) / 16, 128, 0, stream>>>(item_x, wT, xw_i, nI);

    // 5) normalized edge scatter: items->users, users->items
    scatter_edges_kernel<<<(nE + 7) / 8, 256, 0, stream>>>(xw_i, v_s, u_s, rate, cnt_u, h_u, nE);
    scatter_edges_kernel<<<(nE + 7) / 8, 256, 0, stream>>>(xw_u, u_s, v_s, rate, cnt_i, h_v, nE);
}
